// _Joiner_37950331027631
// MI455X (gfx1250) — compile-verified, matrix operand order checked by reading, not run on hardware
//
#include <hip/hip_runtime.h>

typedef _Float16 v16h __attribute__((ext_vector_type(16)));
typedef _Float16 v8h  __attribute__((ext_vector_type(8)));
typedef float    v8f  __attribute__((ext_vector_type(8)));
typedef int      v4i  __attribute__((vector_size(16)));   // matches builtin pointee

// Problem sizes (fixed by the reference)
constexpr int kB = 4;
constexpr int kT = 256;
constexpr int kU = 128;
constexpr int kD = 512;
constexpr int kV = 1024;

// Tiling
constexpr int M_TILE  = 128;                 // == kU: one (b,t) per block in x
constexpr int N_TILE  = 128;                 // waves arranged 2(M) x 4(N)
constexpr int K_CHUNK = 64;                  // staged K slice
constexpr int LDS_PAD = 8;                   // halves; de-conflicts 64 LDS banks
constexpr int LDS_STR = K_CHUNK + LDS_PAD;   // 72 halves per row (144 B)
constexpr int NCHUNK  = kD / K_CHUNK;        // 8

// ---- gfx1250 async global->LDS path (guarded; falls back if unavailable) ----
#if defined(__has_builtin)
#  if __has_builtin(__builtin_amdgcn_global_load_async_to_lds_b128) && \
      __has_builtin(__builtin_amdgcn_s_wait_asynccnt)
#    define HAVE_ASYNC_LDS 1
#  endif
#endif
#ifndef HAVE_ASYNC_LDS
#  define HAVE_ASYNC_LDS 0
#endif

#define GLB_AS __attribute__((address_space(1)))
#define LDS_AS __attribute__((address_space(3)))

#if HAVE_ASYNC_LDS
__device__ __forceinline__ void async_cp16(const _Float16* g, _Float16* l) {
  // flat->AS1: same 64-bit value; flat->AS3: low 32 bits are the LDS offset
  GLB_AS v4i* gp = (GLB_AS v4i*)(uintptr_t)g;
  LDS_AS v4i* lp = (LDS_AS v4i*)(unsigned int)(uintptr_t)l;
  __builtin_amdgcn_global_load_async_to_lds_b128(gp, lp, 0, 0);
}
#endif

__device__ __forceinline__ v16h ld_frag16(const _Float16* p) {
  // p is 16B-aligned by construction (row stride 144B, koff multiple of 32B)
  v8h lo = *(const v8h*)(p);
  v8h hi = *(const v8h*)(p + 8);
  return __builtin_shufflevector(lo, hi, 0,1,2,3,4,5,6,7,8,9,10,11,12,13,14,15);
}

__device__ __forceinline__ v8f wmma_f16(v16h a, v16h b, v8f c) {
  return __builtin_amdgcn_wmma_f32_16x16x32_f16(false, a, false, b,
                                                (short)0, c, false, false);
}

// Pre-convert W (V,D) f32 -> f16 once per launch (2 MB read, trivial)
__global__ __launch_bounds__(256)
void convert_w(const float* __restrict__ W, _Float16* __restrict__ Wh) {
  const int i = (blockIdx.x * 256 + threadIdx.x) * 4;
  const float4 v = *(const float4*)(W + i);
  Wh[i + 0] = (_Float16)v.x;
  Wh[i + 1] = (_Float16)v.y;
  Wh[i + 2] = (_Float16)v.z;
  Wh[i + 3] = (_Float16)v.w;
}

__global__ __launch_bounds__(256)
void joiner_wmma(const float* __restrict__ src,     // (B,T,D)
                 const float* __restrict__ tgt,     // (B,U,D)
                 const float* __restrict__ W,       // (V,D) f32 (fallback)
                 const _Float16* __restrict__ whalf,// (V,D) f16 (async path) or null
                 const float* __restrict__ bias,    // (V)
                 float* __restrict__ out)           // (B,T,U,V)
{
  __shared__ _Float16 Alds[M_TILE * LDS_STR];      // relu(src+tgt) in f16
  __shared__ _Float16 Wl[2][N_TILE * LDS_STR];     // double-buffered W tile

  const int bt     = blockIdx.x;        // b*T + t
  const int b      = bt >> 8;           // / kT
  const int n_base = blockIdx.y * N_TILE;

  const int tid  = threadIdx.x;
  const int wave = tid >> 5;
  const int lane = tid & 31;
  const int wave_m = wave >> 2;         // 0..1 : 64-row half of M tile
  const int wave_n = wave & 3;          // 0..3 : 32-col strip of N tile

  const float* src_row = src + (size_t)bt * kD;
  const float* tgt_blk = tgt + (size_t)b * kU * kD;

  v8f zero = {0.f,0.f,0.f,0.f,0.f,0.f,0.f,0.f};
  v8f acc[4][2];                        // 4 m-tiles x 2 n-tiles per wave
  #pragma unroll
  for (int i = 0; i < 4; ++i) {
    acc[i][0] = zero;
    acc[i][1] = zero;
  }

  // staging assignment: 2 threads per row, each covers 32 consecutive K
  const int srow = tid >> 1;                    // 0..127
  const int skh  = (tid & 1) * (K_CHUNK / 2);   // 0 or 32

  int cur = 0;
  #pragma unroll 1
  for (int kc = 0; kc < NCHUNK; ++kc) {
    const int k0 = kc * K_CHUNK;
    __syncthreads();   // previous iteration's readers are done with LDS

    // ---- stage A = relu(src_row + tgt_row) as f16 (always synchronous) ----
    {
      const float4* tp = (const float4*)(tgt_blk + (size_t)srow * kD + k0 + skh);
      const float4* sp = (const float4*)(src_row + k0 + skh);
      _Float16* ap = &Alds[srow * LDS_STR + skh];
      #pragma unroll
      for (int j = 0; j < (K_CHUNK / 2) / 4; ++j) {
        float4 tv = tp[j];
        float4 sv = sp[j];
        ap[4*j+0] = (_Float16)fmaxf(sv.x + tv.x, 0.0f);
        ap[4*j+1] = (_Float16)fmaxf(sv.y + tv.y, 0.0f);
        ap[4*j+2] = (_Float16)fmaxf(sv.z + tv.z, 0.0f);
        ap[4*j+3] = (_Float16)fmaxf(sv.w + tv.w, 0.0f);
      }
    }

    // ---- stage W tile ----
    bool did_async = false;
#if HAVE_ASYNC_LDS
    if (whalf) {
      // 1024 x 16B segments per chunk; 4 per thread, lane-contiguous in global
      if (kc == 0) {
        #pragma unroll
        for (int i = 0; i < 4; ++i) {
          const int s = i * 256 + tid, row = s >> 3, seg = s & 7;
          async_cp16(whalf + (size_t)(n_base + row) * kD + 0 * K_CHUNK + seg * 8,
                     &Wl[0][row * LDS_STR + seg * 8]);
        }
      }
      if (kc + 1 < NCHUNK) {
        // prefetch next chunk into the other buffer, overlap with compute below
        #pragma unroll
        for (int i = 0; i < 4; ++i) {
          const int s = i * 256 + tid, row = s >> 3, seg = s & 7;
          async_cp16(whalf + (size_t)(n_base + row) * kD + (kc + 1) * K_CHUNK + seg * 8,
                     &Wl[cur ^ 1][row * LDS_STR + seg * 8]);
        }
        __builtin_amdgcn_s_wait_asynccnt(4);  // allow next-chunk copies in flight
      } else {
        __builtin_amdgcn_s_wait_asynccnt(0);  // drain final chunk
      }
      did_async = true;
    }
#endif
    if (!did_async) {
      const float4* wp = (const float4*)(W + (size_t)(n_base + srow) * kD + k0 + skh);
      _Float16* qp = &Wl[cur][srow * LDS_STR + skh];
      #pragma unroll
      for (int j = 0; j < (K_CHUNK / 2) / 4; ++j) {
        float4 wv = wp[j];
        qp[4*j+0] = (_Float16)wv.x;
        qp[4*j+1] = (_Float16)wv.y;
        qp[4*j+2] = (_Float16)wv.z;
        qp[4*j+3] = (_Float16)wv.w;
      }
    }

    // prefetch next A-chunk rows while the WMMA loop runs
    if (kc + 1 < NCHUNK)
      __builtin_prefetch(tgt_blk + (size_t)srow * kD + k0 + K_CHUNK + skh, 0, 1);

    __syncthreads();

    // ---- WMMA: wave tile 64(M) x 32(N); A-frags reused across 2 n-tiles ----
    const _Float16* wlcur = &Wl[cur][0];
    #pragma unroll
    for (int kk = 0; kk < K_CHUNK / 32; ++kk) {
      const int koff = kk * 32 + (lane >> 4) * 16;   // per-lane-half K split
      v16h bfrag[2];
      #pragma unroll
      for (int nt = 0; nt < 2; ++nt)
        bfrag[nt] = ld_frag16(
            &wlcur[(wave_n * 32 + nt * 16 + (lane & 15)) * LDS_STR + koff]);
      #pragma unroll
      for (int g = 0; g < 2; ++g) {
        v16h af[2];
        #pragma unroll
        for (int t = 0; t < 2; ++t)
          af[t] = ld_frag16(
              &Alds[((wave_m * 4 + g * 2 + t) * 16 + (lane & 15)) * LDS_STR + koff]);
        #pragma unroll
        for (int t = 0; t < 2; ++t) {
          acc[g*2+t][0] = wmma_f16(af[t], bfrag[0], acc[g*2+t][0]);
          acc[g*2+t][1] = wmma_f16(af[t], bfrag[1], acc[g*2+t][1]);
        }
      }
    }
    cur ^= 1;
  }

  // ---- epilogue: bias add + coalesced f32 stores ----
  // C/D layout: VGPR r, lanes 0-15 -> M=r, N=lane; lanes 16-31 -> M=r+8, N=lane-16
  const int n0 = n_base + wave_n * 32 + (lane & 15);
  const float bv0 = bias[n0];
  const float bv1 = bias[n0 + 16];
  const int mhalf = (lane >> 4) * 8;
  #pragma unroll
  for (int mt = 0; mt < 4; ++mt) {
    #pragma unroll
    for (int r = 0; r < 8; ++r) {
      const size_t m = (size_t)bt * kU + wave_m * 64 + mt * 16 + mhalf + r;
      out[m * kV + n0]      = acc[mt][0][r] + bv0;
      out[m * kV + n0 + 16] = acc[mt][1][r] + bv1;
    }
  }
}

extern "C" void kernel_launch(void* const* d_in, const int* in_sizes, int n_in,
                              void* d_out, int out_size, void* d_ws, size_t ws_size,
                              hipStream_t stream) {
  const float* src  = (const float*)d_in[0];
  const int*   slen = (const int*)  d_in[1];
  const float* tgt  = (const float*)d_in[2];
  const int*   tlen = (const int*)  d_in[3];
  const float* W    = (const float*)d_in[4];
  const float* bias = (const float*)d_in[5];
  float* out = (float*)d_out;

  // f16 copy of W in scratch, if the harness gave us enough workspace
  const size_t wh_bytes = (size_t)kV * kD * sizeof(_Float16);
  _Float16* whalf = (ws_size >= wh_bytes) ? (_Float16*)d_ws : nullptr;
  if (whalf)
    convert_w<<<(kV * kD) / (256 * 4), 256, 0, stream>>>(W, whalf);

  dim3 grid(kB * kT, kV / N_TILE);   // (1024, 8)
  joiner_wmma<<<grid, 256, 0, stream>>>(src, tgt, W, whalf, bias, out);

  // Reference returns (out, source_lengths, target_lengths): append the two
  // int32 vectors after the main tensor (bit-copy, graph-capture safe).
  const size_t main_elems = (size_t)kB * kT * kU * kV;
  (void)hipMemcpyAsync(out + main_elems,      slen, kB * sizeof(int),
                       hipMemcpyDeviceToDevice, stream);
  (void)hipMemcpyAsync(out + main_elems + kB, tlen, kB * sizeof(int),
                       hipMemcpyDeviceToDevice, stream);
}
